// _SwinBlock_29377576305029
// MI455X (gfx1250) — compile-verified, matrix operand order checked by reading, not run on hardware
//
#include <hip/hip_runtime.h>
#include <hip/hip_bf16.h>
#include <math.h>

typedef unsigned short u16;
typedef __bf16        v16bf __attribute__((ext_vector_type(16)));
typedef float         v8f   __attribute__((ext_vector_type(8)));
typedef unsigned int  u32x4 __attribute__((ext_vector_type(4)));

union FragU { v16bf v; u32x4 q[2]; };

__device__ __forceinline__ u16 f2b(float f) {
  unsigned u = __float_as_uint(f);
  unsigned r = 0x7FFFu + ((u >> 16) & 1u);   // round-to-nearest-even
  return (u16)((u + r) >> 16);
}

// Load a 16x32 bf16 WMMA fragment (A layout; B uses the same mapping with N=lane%16).
// Per ISA: lanes 0-15 hold K = e (e<8) / e+8 (e>=8); lanes 16-31 add +8.
// => two contiguous 16-byte loads per lane.
__device__ __forceinline__ v16bf load_frag(const u16* p) {
  FragU f;
  f.q[0] = *(const u32x4*)(p);
  f.q[1] = *(const u32x4*)(p + 16);
  return f.v;
}

__device__ __forceinline__ v8f wmma_bf16(v16bf a, v16bf b, v8f c) {
  return __builtin_amdgcn_wmma_f32_16x16x32_bf16(false, a, false, b, (short)0, c, false, false);
}

// Async copy 16B from global to LDS (CDNA5 GLOBAL_LOAD_ASYNC_TO_LDS_B128, ASYNCcnt).
__device__ __forceinline__ void async_copy_b128(unsigned lds_off, const void* gaddr) {
  asm volatile("global_load_async_to_lds_b128 %0, %1, off"
               :: "v"(lds_off), "v"((unsigned long long)gaddr) : "memory");
}
__device__ __forceinline__ void async_wait0() {
  asm volatile("s_wait_asynccnt 0" ::: "memory");
}

// ---------------------------------------------------------------- weight cvt
__global__ void f2bf_kernel(const float* __restrict__ a, u16* __restrict__ o, int n) {
  int i = blockIdx.x * 256 + threadIdx.x;
  if (i < n) o[i] = f2b(a[i]);
}

// ---------------------------------------------------------------- LayerNorm
// MODE 0: natural order (LN2).  MODE 1: read shifted coords, write window-major (LN1).
template <int MODE>
__global__ __launch_bounds__(256) void ln_kernel(const float* __restrict__ x,
                                                 const float* __restrict__ g,
                                                 const float* __restrict__ be,
                                                 u16* __restrict__ out, int ntok) {
  int wv = threadIdx.x >> 5;
  int lane = threadIdx.x & 31;
  int tok = blockIdx.x * 8 + wv;
  if (tok >= ntok) return;
  int src = tok;
  if (MODE == 1) {
    int w = tok / 49, t = tok - w * 49;
    int b = w >> 6, wrem = w & 63;
    int wh = wrem >> 3, ww = wrem & 7;
    int ti = t / 7, tj = t - ti * 7;
    int r = wh * 7 + ti, c = ww * 7 + tj;     // shifted coords
    int i = r + 3; if (i >= 56) i -= 56;      // roll(-3): shifted[r] = x[(r+3)%56]
    int j = c + 3; if (j >= 56) j -= 56;
    src = b * 3136 + i * 56 + j;
  }
  const float* xr = x + (size_t)src * 192;
  float v[6];
  float s = 0.f, s2 = 0.f;
#pragma unroll
  for (int q = 0; q < 6; ++q) {
    float t = xr[lane + 32 * q];
    v[q] = t; s += t; s2 += t * t;
  }
#pragma unroll
  for (int off = 16; off > 0; off >>= 1) {
    s  += __shfl_xor(s, off, 32);
    s2 += __shfl_xor(s2, off, 32);
  }
  float mean = s * (1.f / 192.f);
  float var  = s2 * (1.f / 192.f) - mean * mean;
  float rs   = rsqrtf(var + 1e-5f);
  u16* orow = out + (size_t)tok * 192;
#pragma unroll
  for (int q = 0; q < 6; ++q) {
    int col = lane + 32 * q;
    orow[col] = f2b((v[q] - mean) * rs * g[col] + be[col]);
  }
}

// ---------------------------------------------------------------- GEMM (bf16 WMMA)
// C[M,N] = A[M,K](bf16) @ W[N,K]^T(bf16) + bias, with fused epilogues.
// Each wave owns a 32(M) x 64(N) tile: 2 A-frags amortize 4 B-frags per k-step.
enum { EP_BF16 = 0, EP_GELU = 1, EP_OUTPROJ = 2, EP_MLP2 = 3 };

template <int EP>
__global__ __launch_bounds__(256) void gemm_kernel(const u16* __restrict__ A,
                                                   const u16* __restrict__ W,
                                                   const float* __restrict__ bias,
                                                   int M, int N, int K,
                                                   u16* __restrict__ outb,
                                                   float* __restrict__ outf,
                                                   const float* __restrict__ res) {
  int lane = threadIdx.x & 31;
  int wgid = blockIdx.x * 8 + (threadIdx.x >> 5);
  int ntg = N >> 6;
  int total = (M >> 5) * ntg;
  if (wgid >= total) return;
  int mt = wgid / ntg, ng = wgid - mt * ntg;
  int mbase = mt << 5, nbase = ng << 6;
  int hi = (lane & 16) ? 8 : 0;
  int nl = lane & 15;
  const u16* arow0 = A + (size_t)(mbase + nl) * K + hi;
  const u16* arow1 = arow0 + (size_t)16 * K;
  const u16* brow  = W + (size_t)(nbase + nl) * K + hi;
  v8f acc[2][4] = {};
  int ks = K >> 5;
  for (int kt = 0; kt < ks; ++kt) {
    // prefetch the streaming A operand two k-steps ahead (global_prefetch_b8)
    __builtin_prefetch((const void*)(arow0 + kt * 32 + 64), 0, 1);
    __builtin_prefetch((const void*)(arow1 + kt * 32 + 64), 0, 1);
    v16bf a0 = load_frag(arow0 + kt * 32);
    v16bf a1 = load_frag(arow1 + kt * 32);
#pragma unroll
    for (int nt = 0; nt < 4; ++nt) {
      v16bf b = load_frag(brow + (size_t)(nt * 16) * K + kt * 32);
      acc[0][nt] = wmma_bf16(a0, b, acc[0][nt]);
      acc[1][nt] = wmma_bf16(a1, b, acc[1][nt]);
    }
  }
#pragma unroll
  for (int half = 0; half < 2; ++half) {
#pragma unroll
    for (int r = 0; r < 8; ++r) {
      int m = mbase + half * 16 + r + hi;    // D layout: lanes>=16 own M = r+8
      size_t drow = 0;
      if (EP == EP_OUTPROJ) {                // window-major row -> unshifted token
        int w = m / 49, t = m - w * 49;
        int b = w >> 6, wrem = w & 63;
        int wh = wrem >> 3, ww = wrem & 7;
        int ti = t / 7, tj = t - ti * 7;
        int rr = wh * 7 + ti, cc = ww * 7 + tj;
        int i = rr + 3; if (i >= 56) i -= 56;
        int j = cc + 3; if (j >= 56) j -= 56;
        drow = (size_t)(b * 3136 + i * 56 + j) * 192;
      }
#pragma unroll
      for (int nt = 0; nt < 4; ++nt) {
        int n = nbase + nt * 16 + nl;
        float c = acc[half][nt][r] + bias[n];
        if (EP == EP_BF16) {
          outb[(size_t)m * N + n] = f2b(c);
        } else if (EP == EP_GELU) {
          float gl = 0.5f * c * (1.f + erff(c * 0.70710678118654752f));
          outb[(size_t)m * N + n] = f2b(gl);
        } else if (EP == EP_OUTPROJ) {
          outf[drow + n] = c + res[drow + n];          // + residual (original x)
        } else {                                        // EP_MLP2
          size_t di = (size_t)m * N + n;
          outf[di] = res[di] + c;                       // y + MLP(y)
        }
      }
    }
  }
}

// ---------------------------------------------------------------- attention
// One block (4 waves) per (window, head): S = QK^T/sqrt(dh), softmax, P@V.
__global__ __launch_bounds__(128) void attn_kernel(const u16* __restrict__ qkv,
                                                   u16* __restrict__ o) {
  int w = blockIdx.x / 6, h = blockIdx.x - w * 6;
  __shared__ __align__(16) u16 Qs[64 * 32];
  __shared__ __align__(16) u16 Ks[64 * 32];
  __shared__ __align__(16) u16 Vt[32 * 64];   // V transposed: Vt[d][t]
  __shared__ float S[64 * 64];
  __shared__ __align__(16) u16 P[64 * 64];
  int tid = threadIdx.x;
  // Q/K staged via async global->LDS b128 copies (16B per token-quarter).
  for (int idx = tid; idx < 49 * 4; idx += 128) {
    int t = idx >> 2, c = idx & 3;
    const u16* qsrc = qkv + (size_t)(w * 49 + t) * 576 + h * 32 + c * 8;
    async_copy_b128((unsigned)(size_t)&Qs[t * 32 + c * 8], qsrc);
    async_copy_b128((unsigned)(size_t)&Ks[t * 32 + c * 8], qsrc + 192);
  }
  // V transposed (scalar scatter); padded columns t>=49 MUST be zero.
  for (int idx = tid; idx < 64 * 32; idx += 128) {
    int t = idx >> 5, d = idx & 31;
    u16 vv = 0;
    if (t < 49) vv = qkv[(size_t)(w * 49 + t) * 576 + 384 + h * 32 + d];
    Vt[d * 64 + t] = vv;
  }
  async_wait0();
  __syncthreads();
  int lane = tid & 31, wv = tid >> 5;
  int hi = (lane & 16) ? 8 : 0;
  int nl = lane & 15;
  int mbase = wv * 16;
  {  // S = Q @ K^T : K-dim = 32 = exactly one bf16 WMMA per 16x16 tile
    v16bf a = load_frag(&Qs[(mbase + nl) * 32 + hi]);
#pragma unroll
    for (int nt = 0; nt < 4; ++nt) {
      v16bf b = load_frag(&Ks[(nt * 16 + nl) * 32 + hi]);
      v8f acc = {};
      acc = wmma_bf16(a, b, acc);
#pragma unroll
      for (int r = 0; r < 8; ++r)
        S[(mbase + r + hi) * 64 + nt * 16 + nl] = acc[r];
    }
  }
  __syncthreads();
  if (tid < 64) {
    int m = tid;
    if (m < 49) {
      const float scale = 0.17677669529663687f;  // 1/sqrt(32)
      float mx = -3.0e38f;
      for (int n = 0; n < 49; ++n) mx = fmaxf(mx, S[m * 64 + n] * scale);
      float sum = 0.f;
      for (int n = 0; n < 49; ++n) {
        float e = expf(S[m * 64 + n] * scale - mx);
        sum += e;
        S[m * 64 + n] = e;
      }
      float inv = 1.f / sum;
      for (int n = 0; n < 64; ++n)
        P[m * 64 + n] = (n < 49) ? f2b(S[m * 64 + n] * inv) : (u16)0;
    } else {
      for (int n = 0; n < 64; ++n) P[m * 64 + n] = 0;  // padded Q rows
    }
  }
  __syncthreads();
  v8f acc0 = {}, acc1 = {};
#pragma unroll
  for (int k0 = 0; k0 < 64; k0 += 32) {   // O = P @ V, K-dim 64 (49 padded)
    v16bf a  = load_frag(&P[(mbase + nl) * 64 + k0 + hi]);
    v16bf b0 = load_frag(&Vt[(0  + nl) * 64 + k0 + hi]);
    v16bf b1 = load_frag(&Vt[(16 + nl) * 64 + k0 + hi]);
    acc0 = wmma_bf16(a, b0, acc0);
    acc1 = wmma_bf16(a, b1, acc1);
  }
#pragma unroll
  for (int r = 0; r < 8; ++r) {
    int t = mbase + r + hi;
    if (t < 49) {
      size_t row = (size_t)(w * 49 + t) * 192 + h * 32;
      o[row + nl]      = f2b(acc0[r]);
      o[row + 16 + nl] = f2b(acc1[r]);
    }
  }
}

// ---------------------------------------------------------------- host
extern "C" void kernel_launch(void* const* d_in, const int* in_sizes, int n_in,
                              void* d_out, int out_size, void* d_ws, size_t ws_size,
                              hipStream_t stream) {
  const float* x    = (const float*)d_in[0];
  const float* ln1g = (const float*)d_in[1];
  const float* ln1b = (const float*)d_in[2];
  const float* ln2g = (const float*)d_in[3];
  const float* ln2b = (const float*)d_in[4];
  const float* wqkv = (const float*)d_in[5];
  const float* bqkv = (const float*)d_in[6];
  const float* wout = (const float*)d_in[7];
  const float* bout = (const float*)d_in[8];
  const float* w1   = (const float*)d_in[9];
  const float* b1   = (const float*)d_in[10];
  const float* w2   = (const float*)d_in[11];
  const float* b2   = (const float*)d_in[12];
  float* out = (float*)d_out;
  (void)n_in; (void)out_size; (void)ws_size;

  int Bsz = in_sizes[0] / (56 * 56 * 192);
  int tokens = Bsz * 56 * 56;       // == windows*49
  int wins = Bsz * 64;

  char* ws = (char*)d_ws;
  size_t off = 0;
  auto alloc = [&](size_t bytes) {
    size_t o = off;
    off = (off + bytes + 255) & ~(size_t)255;
    return o;
  };
  size_t o_wq  = alloc((size_t)576 * 192 * 2);
  size_t o_wo  = alloc((size_t)192 * 192 * 2);
  size_t o_w1  = alloc((size_t)384 * 192 * 2);
  size_t o_w2  = alloc((size_t)192 * 384 * 2);
  size_t o_xw  = alloc((size_t)tokens * 192 * 2);  // LN1 windows; reused as LN2 out
  size_t o_qkv = alloc((size_t)tokens * 576 * 2);  // reused as MLP hidden (tokens*384*2)
  size_t o_att = alloc((size_t)tokens * 192 * 2);
  size_t o_y   = alloc((size_t)tokens * 192 * 4);

  u16* wq_b = (u16*)(ws + o_wq);
  u16* wo_b = (u16*)(ws + o_wo);
  u16* w1_b = (u16*)(ws + o_w1);
  u16* w2_b = (u16*)(ws + o_w2);
  u16* xw   = (u16*)(ws + o_xw);
  u16* qkvb = (u16*)(ws + o_qkv);
  u16* attb = (u16*)(ws + o_att);
  float* yf = (float*)(ws + o_y);

  // weights -> bf16
  f2bf_kernel<<<(576 * 192 + 255) / 256, 256, 0, stream>>>(wqkv, wq_b, 576 * 192);
  f2bf_kernel<<<(192 * 192 + 255) / 256, 256, 0, stream>>>(wout, wo_b, 192 * 192);
  f2bf_kernel<<<(384 * 192 + 255) / 256, 256, 0, stream>>>(w1,   w1_b, 384 * 192);
  f2bf_kernel<<<(192 * 384 + 255) / 256, 256, 0, stream>>>(w2,   w2_b, 192 * 384);

  // shift + LN1 + window partition
  ln_kernel<1><<<(tokens + 7) / 8, 256, 0, stream>>>(x, ln1g, ln1b, xw, tokens);

  // QKV projection [M,192] x [192,576]^T
  {
    int total = (tokens >> 5) * (576 >> 6);
    gemm_kernel<EP_BF16><<<(total + 7) / 8, 256, 0, stream>>>(
        xw, wq_b, bqkv, tokens, 576, 192, qkvb, nullptr, nullptr);
  }

  // fused window attention
  attn_kernel<<<wins * 6, 128, 0, stream>>>(qkvb, attb);

  // out projection + window-reverse + unshift + residual -> y (fp32)
  {
    int total = (tokens >> 5) * (192 >> 6);
    gemm_kernel<EP_OUTPROJ><<<(total + 7) / 8, 256, 0, stream>>>(
        attb, wo_b, bout, tokens, 192, 192, nullptr, yf, x);
  }

  // LN2
  ln_kernel<0><<<(tokens + 7) / 8, 256, 0, stream>>>(yf, ln2g, ln2b, xw, tokens);

  // MLP1 + exact GELU
  {
    int total = (tokens >> 5) * (384 >> 6);
    gemm_kernel<EP_GELU><<<(total + 7) / 8, 256, 0, stream>>>(
        xw, w1_b, b1, tokens, 384, 192, qkvb, nullptr, nullptr);
  }

  // MLP2 + residual -> out (fp32)
  {
    int total = (tokens >> 5) * (192 >> 6);
    gemm_kernel<EP_MLP2><<<(total + 7) / 8, 256, 0, stream>>>(
        qkvb, w2_b, b2, tokens, 192, 384, nullptr, out, yf);
  }
}
